// FastTopoMambaFoundation_30348238913552
// MI455X (gfx1250) — matrix operand order repaired and checked by reading, not compiled, not run on hardware
//
#include <hip/hip_runtime.h>
#include <hip/hip_bf16.h>

typedef _Float16 v16h __attribute__((ext_vector_type(16)));
typedef _Float16 v8h  __attribute__((ext_vector_type(8)));
typedef float    v8f  __attribute__((ext_vector_type(8)));

#define BB 4
#define RR 256
#define PP 64
#define DP 384
#define DR 512
#define NN 48
#define RK 24
#define NG 4              // scan n-groups
#define NC (NN / NG)      // 12 states per group

__device__ __forceinline__ float geluf(float x) {
    return 0.5f * x * (1.0f + erff(x * 0.7071067811865475f));
}
__device__ __forceinline__ float siluf(float x) {
    return x / (1.0f + expf(-x));
}
__device__ __forceinline__ float softplusf(float x) {
    return (x > 20.0f) ? x : log1pf(expf(x));
}
__device__ __forceinline__ float nan2num(float v) {
    if (isnan(v)) return 0.0f;
    if (isinf(v)) return v > 0.0f ? 1.0f : -1.0f;
    return v;
}

// fixed-order block reduction (deterministic)
__device__ __forceinline__ float blockReduceSum(float v, float* scratch) {
    int t = threadIdx.x;
    scratch[t] = v;
    __syncthreads();
    for (int s = 128; s > 0; s >>= 1) {
        if (t < s) scratch[t] += scratch[t + s];
        __syncthreads();
    }
    float r = scratch[0];
    __syncthreads();
    return r;
}

// ---------------- prep: gate_w1 (384x512 f32) -> w1t (512x384 f16) ----------------
__global__ void prep_w1t_kernel(const float* __restrict__ w1, _Float16* __restrict__ w1t) {
    int idx = blockIdx.x * 256 + threadIdx.x;
    if (idx < DP * DR) {
        int k = idx / DR;
        int n = idx % DR;
        w1t[n * DP + k] = (_Float16)w1[idx];
    }
}

// ---------------- kernel 1: per-region gate GEMM (WMMA) + pool + proj tail ----------------
__global__ __launch_bounds__(256) void region_kernel(
    const float* __restrict__ pt,        // patch_tokens (B,R,P,DP)
    const int*   __restrict__ pmask,     // (B,R,P)
    const float* __restrict__ lp,        // local_pos (P,DP)
    const float* __restrict__ gln_w, const float* __restrict__ gln_b,
    const _Float16* __restrict__ w1t,    // (DR, DP) f16
    const float* __restrict__ gb1,       // (DR)
    const float* __restrict__ gw2,       // (DR)
    const float* __restrict__ gb2,       // (1)
    const float* __restrict__ pln1_w, const float* __restrict__ pln1_b,
    const float* __restrict__ pw1, const float* __restrict__ pb1,   // (DP,DR),(DR)
    const float* __restrict__ pw2, const float* __restrict__ pb2,   // (DR,DR),(DR)
    const float* __restrict__ pln2_w, const float* __restrict__ pln2_b,
    const float* __restrict__ in_w,      // (DR, 2DR)
    const float* __restrict__ xw,        // (DR, 120)
    const float* __restrict__ dtw,       // (RK, DR)
    const float* __restrict__ dtb,       // (DR)
    float* __restrict__ ws_region, float* __restrict__ ws_u, float* __restrict__ ws_z,
    float* __restrict__ ws_delta, float* __restrict__ ws_B, float* __restrict__ ws_C)
{
    __shared__ __align__(16) _Float16 a16[PP * DP];   // 49152 B
    __shared__ float smem[3328];                      // 13312 B

    float* attn   = smem;          // [0,64)
    float* logits = smem + 64;     // [64,128)
    float* lpart  = smem + 128;    // [128,640)  8 waves x 64 rows
    float* scratch= smem + 640;    // [640,896)
    float* pooled = smem + 896;    // [896,1280)
    float* phat   = smem + 1280;   // [1280,1664)
    float* p1     = smem + 1664;   // [1664,2176)
    float* pr     = smem + 2176;   // [2176,2688)  (later: region)
    float* uvec   = smem + 2688;   // [2688,3200)
    float* prm    = smem + 3200;   // [3200,3320)
    float* redS   = smem + 128;    // phase-A overlays lpart
    float* redQ   = smem + 384;

    const int tid = threadIdx.x;
    const int bid = blockIdx.x;          // (b,r) flat, 0..1023
    const size_t tokbase = (size_t)bid * PP * DP;

    // ---- Phase A: load x = tokens + pos, LayerNorm per row -> a16 (f16) ----
    {
        int row = tid >> 2;          // 0..63
        int q   = tid & 3;           // 0..3, 96 cols each
        const float* xr = pt + tokbase + (size_t)row * DP;
        const float* lr = lp + row * DP;
        float s = 0.f, sq = 0.f;
        for (int c = q * 96; c < q * 96 + 96; ++c) {
            float v = xr[c] + lr[c];
            s += v; sq += v * v;
        }
        redS[row * 4 + q] = s;
        redQ[row * 4 + q] = sq;
        __syncthreads();
        float sum = redS[row*4+0] + redS[row*4+1] + redS[row*4+2] + redS[row*4+3];
        float ssq = redQ[row*4+0] + redQ[row*4+1] + redQ[row*4+2] + redQ[row*4+3];
        float mean = sum * (1.0f / DP);
        float var  = ssq * (1.0f / DP) - mean * mean;
        float inv  = rsqrtf(var + 1e-5f);
        for (int c = q * 96; c < q * 96 + 96; ++c) {
            float v = xr[c] + lr[c];
            a16[row * DP + c] = (_Float16)(((v - mean) * inv) * gln_w[c] + gln_b[c]);
        }
        __syncthreads();
    }

    // ---- Phase B: GEMM (64x384)@(384x512) via WMMA f16, fused gelu * gate_w2 ----
    const int lane  = tid & 31;
    const int wv    = tid >> 5;      // 0..7: owns N slice [wv*64, wv*64+64)
    const int lhalf = lane >> 4;     // 0/1
    const int l15   = lane & 15;

    v8f acc[4][4];
    #pragma unroll
    for (int mt = 0; mt < 4; ++mt)
        #pragma unroll
        for (int nt = 0; nt < 4; ++nt)
            acc[mt][nt] = (v8f){0.f,0.f,0.f,0.f,0.f,0.f,0.f,0.f};

    for (int k0 = 0; k0 < DP; k0 += 32) {
        v16h afr[4];
        #pragma unroll
        for (int mt = 0; mt < 4; ++mt) {
            int row = mt * 16 + l15;
            int ka  = k0 + lhalf * 8;
            v8h lo = *(const v8h*)&a16[row * DP + ka];
            v8h hi = *(const v8h*)&a16[row * DP + ka + 16];
            afr[mt] = __builtin_shufflevector(lo, hi, 0,1,2,3,4,5,6,7,8,9,10,11,12,13,14,15);
        }
        #pragma unroll
        for (int nt = 0; nt < 4; ++nt) {
            int n  = wv * 64 + nt * 16 + l15;
            int kb = k0 + lhalf * 8;
            const v8h* pb = (const v8h*)&w1t[n * DP + kb];
            v16h bfr = __builtin_shufflevector(pb[0], pb[2], 0,1,2,3,4,5,6,7,8,9,10,11,12,13,14,15);
            #pragma unroll
            for (int mt = 0; mt < 4; ++mt)
                acc[mt][nt] = __builtin_amdgcn_wmma_f32_16x16x32_f16(
                    false, afr[mt], false, bfr, (short)0, acc[mt][nt], false, false);
        }
    }

    // fused epilogue: gelu(acc + b1) * w2, partial logits per row
    float part[4][8];
    #pragma unroll
    for (int mt = 0; mt < 4; ++mt)
        #pragma unroll
        for (int v = 0; v < 8; ++v) part[mt][v] = 0.f;

    #pragma unroll
    for (int nt = 0; nt < 4; ++nt) {
        int n = wv * 64 + nt * 16 + l15;
        float b1v = gb1[n];
        float w2v = gw2[n];
        #pragma unroll
        for (int mt = 0; mt < 4; ++mt)
            #pragma unroll
            for (int v = 0; v < 8; ++v)
                part[mt][v] += geluf(acc[mt][nt][v] + b1v) * w2v;
    }
    // reduce across the 16 columns (lanes within each 16-lane half)
    #pragma unroll
    for (int mt = 0; mt < 4; ++mt)
        #pragma unroll
        for (int v = 0; v < 8; ++v) {
            float x = part[mt][v];
            x += __shfl_xor(x, 1, 32);
            x += __shfl_xor(x, 2, 32);
            x += __shfl_xor(x, 4, 32);
            x += __shfl_xor(x, 8, 32);
            part[mt][v] = x;
        }
    if (l15 == 0) {
        #pragma unroll
        for (int mt = 0; mt < 4; ++mt)
            #pragma unroll
            for (int v = 0; v < 8; ++v) {
                int row = mt * 16 + lhalf * 8 + v;
                lpart[wv * 64 + row] = part[mt][v];
            }
    }
    __syncthreads();
    if (tid < 64) {
        float s = gb2[0];
        for (int w = 0; w < 8; ++w) s += lpart[w * 64 + tid];
        logits[tid] = s;
    }
    __syncthreads();

    // ---- masked softmax (thread 0, 64 elements) ----
    if (tid == 0) {
        const int* pm = pmask + bid * PP;
        float mx = -3.402823466e38f;
        for (int p = 0; p < PP; ++p)
            if (pm[p]) mx = fmaxf(mx, logits[p]);
        float s = 0.f;
        for (int p = 0; p < PP; ++p) {
            float e = pm[p] ? expf(logits[p] - mx) : 0.f;
            attn[p] = e; s += e;
        }
        float inv = 1.0f / fmaxf(s, 1e-6f);
        for (int p = 0; p < PP; ++p) attn[p] *= inv;
    }
    __syncthreads();

    // ---- Phase C: pooled[d] = sum_p attn[p] * x[p][d] ----
    for (int d = tid; d < DP; d += 256) {
        float s = 0.f;
        for (int p = 0; p < PP; ++p)
            s += attn[p] * (pt[tokbase + (size_t)p * DP + d] + lp[p * DP + d]);
        pooled[d] = s;
    }
    __syncthreads();

    // ---- Phase D: per-region proj MLP tail ----
    // LN1 over pooled(384)
    {
        float v0 = pooled[tid] + ((tid < 128) ? pooled[tid + 256] : 0.f);
        float mean = blockReduceSum(v0, scratch) * (1.0f / DP);
        float d0 = pooled[tid] - mean;
        float d1 = (tid < 128) ? (pooled[tid + 256] - mean) : 0.f;
        float var = blockReduceSum(d0 * d0 + d1 * d1, scratch) * (1.0f / DP);
        float inv = rsqrtf(var + 1e-5f);
        phat[tid] = (pooled[tid] - mean) * inv * pln1_w[tid] + pln1_b[tid];
        if (tid < 128) {
            int j = tid + 256;
            phat[j] = (pooled[j] - mean) * inv * pln1_w[j] + pln1_b[j];
        }
        __syncthreads();
    }
    // p1 = gelu(phat @ proj_w1 + b1)
    for (int c = 0; c < 2; ++c) {
        int j = tid + c * 256;
        float s = pb1[j];
        for (int k = 0; k < DP; ++k) s += phat[k] * pw1[k * DR + j];
        p1[j] = geluf(s);
    }
    __syncthreads();
    // p2 = p1 @ proj_w2 + b2 ; LN2 ; nan_to_num
    for (int c = 0; c < 2; ++c) {
        int j = tid + c * 256;
        float s = pb2[j];
        for (int k = 0; k < DR; ++k) s += p1[k] * pw2[k * DR + j];
        pr[j] = s;
    }
    __syncthreads();
    {
        float mean = blockReduceSum(pr[tid] + pr[tid + 256], scratch) * (1.0f / DR);
        float d0 = pr[tid] - mean, d1 = pr[tid + 256] - mean;
        float var = blockReduceSum(d0 * d0 + d1 * d1, scratch) * (1.0f / DR);
        float inv = rsqrtf(var + 1e-5f);
        float a = nan2num(d0 * inv * pln2_w[tid] + pln2_b[tid]);
        float b = nan2num(d1 * inv * pln2_w[tid + 256] + pln2_b[tid + 256]);
        __syncthreads();
        pr[tid] = a; pr[tid + 256] = b;
        __syncthreads();
    }
    // region = l2norm(pr)
    {
        float nrm2 = blockReduceSum(pr[tid] * pr[tid] + pr[tid + 256] * pr[tid + 256], scratch);
        float invn = 1.0f / fmaxf(sqrtf(nrm2), 1e-6f);
        float a = pr[tid] * invn, b = pr[tid + 256] * invn;
        __syncthreads();
        pr[tid] = a; pr[tid + 256] = b;
        ws_region[(size_t)bid * DR + tid] = a;
        ws_region[(size_t)bid * DR + tid + 256] = b;
        __syncthreads();
    }
    // xz = region @ in_proj_w (512 -> 1024); u,z
    for (int c = 0; c < 4; ++c) {
        int j = tid + c * 256;
        float s = 0.f;
        for (int k = 0; k < DR; ++k) s += pr[k] * in_w[k * 2 * DR + j];
        if (j < DR) { uvec[j] = s; ws_u[(size_t)bid * DR + j] = s; }
        else        { ws_z[(size_t)bid * DR + (j - DR)] = s; }
    }
    __syncthreads();
    // prm = u @ x_proj_w (512 -> 120)
    if (tid < RK + 2 * NN) {
        float s = 0.f;
        for (int k = 0; k < DR; ++k) s += uvec[k] * xw[k * (RK + 2 * NN) + tid];
        prm[tid] = s;
    }
    __syncthreads();
    if (tid < NN) {
        ws_B[(size_t)bid * NN + tid] = prm[RK + tid];
        ws_C[(size_t)bid * NN + tid] = prm[RK + NN + tid];
    }
    // delta = min(softplus(dt_raw @ dt_w + dt_b), 10)
    for (int c = 0; c < 2; ++c) {
        int j = tid + c * 256;
        float s = dtb[j];
        for (int k = 0; k < RK; ++k) s += prm[k] * dtw[k * DR + j];
        ws_delta[(size_t)bid * DR + j] = fminf(softplusf(s), 10.0f);
    }
}

// ---------------- kernel 2: SSM scan over R, thread = (b,d,ngroup of 12 states) ----------------
__global__ __launch_bounds__(256) void scan_kernel(
    const float* __restrict__ ws_delta, const float* __restrict__ ws_u,
    const float* __restrict__ ws_B, const float* __restrict__ ws_C,
    const float* __restrict__ Alog,
    float* __restrict__ ws_ypart)       // (NG, B*R, DR)
{
    int g = blockIdx.x * 256 + threadIdx.x;   // 0 .. B*DR*NG-1
    int d  = g & 511;
    int t  = g >> 9;
    int b  = t & 3;
    int ng = t >> 2;                          // 0..NG-1
    int n0 = ng * NC;

    float Aneg[NC], h[NC];
    #pragma unroll
    for (int n = 0; n < NC; ++n) {
        Aneg[n] = -fminf(expf(Alog[d * NN + n0 + n]), 10000.0f);
        h[n] = 0.f;
    }
    float* ypart = ws_ypart + (size_t)ng * BB * RR * DR;
    for (int r = 0; r < RR; ++r) {
        int idx = b * RR + r;
        float dv = ws_delta[(size_t)idx * DR + d];
        float uv = ws_u[(size_t)idx * DR + d];
        float du = dv * uv;
        float acc = 0.f;
        #pragma unroll
        for (int n = 0; n < NC; ++n) {
            float tt = fminf(fmaxf(dv * Aneg[n], -30.0f), 30.0f);
            float dA = __expf(tt);
            h[n] = dA * h[n] + du * ws_B[(size_t)idx * NN + n0 + n];
            acc += ws_C[(size_t)idx * NN + n0 + n] * h[n];
        }
        ypart[(size_t)idx * DR + d] = acc;
    }
}

// ---------------- kernel 3: y = (y_seq*silu(z)) @ out_proj ; out = LN(region + y) ; rt f16 ----------------
__global__ __launch_bounds__(256) void out_kernel(
    const float* __restrict__ ws_ypart, const float* __restrict__ ws_u,
    const float* __restrict__ ws_z, const float* __restrict__ ws_region,
    const float* __restrict__ Dp,
    const float* __restrict__ outw,     // (DR,DR)
    const float* __restrict__ sln_w, const float* __restrict__ sln_b,
    float* __restrict__ d_out, _Float16* __restrict__ rt16)
{
    __shared__ float ym[DR];
    __shared__ float vrow[DR];
    __shared__ float scratch[256];
    int tid = threadIdx.x;
    size_t idx = blockIdx.x;
    const size_t gstride = (size_t)BB * RR * DR;

    for (int c = 0; c < 2; ++c) {
        int j = tid + c * 256;
        float ys = Dp[j] * ws_u[idx * DR + j];
        #pragma unroll
        for (int ng = 0; ng < NG; ++ng)
            ys += ws_ypart[(size_t)ng * gstride + idx * DR + j];
        float zv = ws_z[idx * DR + j];
        ym[j] = ys * siluf(zv);
    }
    __syncthreads();
    for (int c = 0; c < 2; ++c) {
        int j = tid + c * 256;
        float s = 0.f;
        for (int k = 0; k < DR; ++k) s += ym[k] * outw[k * DR + j];
        vrow[j] = ws_region[idx * DR + j] + nan2num(s);
    }
    __syncthreads();
    float mean = blockReduceSum(vrow[tid] + vrow[tid + 256], scratch) * (1.0f / DR);
    float d0 = vrow[tid] - mean, d1 = vrow[tid + 256] - mean;
    float var = blockReduceSum(d0 * d0 + d1 * d1, scratch) * (1.0f / DR);
    float inv = rsqrtf(var + 1e-5f);
    float o0 = d0 * inv * sln_w[tid] + sln_b[tid];
    float o1 = d1 * inv * sln_w[tid + 256] + sln_b[tid + 256];
    d_out[idx * DR + tid] = o0;
    d_out[idx * DR + tid + 256] = o1;
    __syncthreads();
    vrow[tid] = o0; vrow[tid + 256] = o1;
    __syncthreads();
    float nrm2 = blockReduceSum(o0 * o0 + o1 * o1, scratch);
    float invn = 1.0f / fmaxf(sqrtf(nrm2), 1e-6f);
    rt16[idx * DR + tid]       = (_Float16)(o0 * invn);
    rt16[idx * DR + tid + 256] = (_Float16)(o1 * invn);
}

// ---------------- kernel 4: sim tiles via WMMA; masked partial loss sums ----------------
__global__ __launch_bounds__(32) void sim_kernel(
    const _Float16* __restrict__ rt16,
    const int* __restrict__ nearm, const int* __restrict__ farm,
    float* __restrict__ partials)
{
    int bid = blockIdx.x;              // b*256 + mt*16 + nt
    int b  = bid >> 8;
    int mt = (bid >> 4) & 15;
    int nt = bid & 15;
    int m0 = mt * 16, n0 = nt * 16;

    int lane  = threadIdx.x;
    int lhalf = lane >> 4;
    int l15   = lane & 15;

    v8f acc = (v8f){0.f,0.f,0.f,0.f,0.f,0.f,0.f,0.f};
    const _Float16* base = rt16 + (size_t)b * RR * DR;

    for (int k0 = 0; k0 < DR; k0 += 32) {
        // A: rows m0..m0+15 of rt
        int rowA = m0 + l15;
        int ka   = k0 + lhalf * 8;
        v8h alo = *(const v8h*)&base[(size_t)rowA * DR + ka];
        v8h ahi = *(const v8h*)&base[(size_t)rowA * DR + ka + 16];
        v16h afr = __builtin_shufflevector(alo, ahi, 0,1,2,3,4,5,6,7,8,9,10,11,12,13,14,15);
        // B: columns = rows n0..n0+15 of rt (rt^T)
        int rowB = n0 + l15;
        int kb   = k0 + lhalf * 8;
        const v8h* pb = (const v8h*)&base[(size_t)rowB * DR + kb];
        v16h bfr = __builtin_shufflevector(pb[0], pb[2], 0,1,2,3,4,5,6,7,8,9,10,11,12,13,14,15);
        acc = __builtin_amdgcn_wmma_f32_16x16x32_f16(false, afr, false, bfr, (short)0, acc, false, false);
    }

    float ns = 0.f, fs = 0.f;
    #pragma unroll
    for (int v = 0; v < 8; ++v) {
        int r = m0 + lhalf * 8 + v;
        int s = n0 + l15;
        float sim = acc[v];
        if (nearm[r * RR + s]) ns += (1.0f - sim);
        if (farm[r * RR + s])  fs += fmaxf(sim - 0.2f, 0.0f);
    }
    ns += __shfl_xor(ns, 16, 32); fs += __shfl_xor(fs, 16, 32);
    ns += __shfl_xor(ns, 8, 32);  fs += __shfl_xor(fs, 8, 32);
    ns += __shfl_xor(ns, 4, 32);  fs += __shfl_xor(fs, 4, 32);
    ns += __shfl_xor(ns, 2, 32);  fs += __shfl_xor(fs, 2, 32);
    ns += __shfl_xor(ns, 1, 32);  fs += __shfl_xor(fs, 1, 32);
    if (lane == 0) {
        partials[bid * 2]     = ns;
        partials[bid * 2 + 1] = fs;
    }
}

// ---------------- kernel 5: final loss (fixed-order reduction) ----------------
__global__ __launch_bounds__(256) void loss_kernel(
    const float* __restrict__ partials,
    const int* __restrict__ nearm, const int* __restrict__ farm,
    float* __restrict__ out_loss)
{
    __shared__ float sA[256], sB[256], sC[256], sD[256];
    int t = threadIdx.x;
    float ns = 0.f, fs = 0.f;
    for (int i = t; i < 1024; i += 256) { ns += partials[2 * i]; fs += partials[2 * i + 1]; }
    float nc = 0.f, fc = 0.f;
    for (int i = t; i < RR * RR; i += 256) { nc += (nearm[i] != 0); fc += (farm[i] != 0); }
    sA[t] = ns; sB[t] = fs; sC[t] = nc; sD[t] = fc;
    __syncthreads();
    for (int s = 128; s > 0; s >>= 1) {
        if (t < s) { sA[t]+=sA[t+s]; sB[t]+=sB[t+s]; sC[t]+=sC[t+s]; sD[t]+=sD[t+s]; }
        __syncthreads();
    }
    if (t == 0)
        out_loss[0] = sA[0] / fmaxf(sC[0], 1.0f) + sB[0] / fmaxf(sD[0], 1.0f);
}

extern "C" void kernel_launch(void* const* d_in, const int* in_sizes, int n_in,
                              void* d_out, int out_size, void* d_ws, size_t ws_size,
                              hipStream_t stream) {
    const float* pt     = (const float*)d_in[0];
    const int*   pmask  = (const int*)  d_in[1];
    const int*   nearm  = (const int*)  d_in[2];
    const int*   farm   = (const int*)  d_in[3];
    const float* lp     = (const float*)d_in[4];
    const float* gln_w  = (const float*)d_in[5];
    const float* gln_b  = (const float*)d_in[6];
    const float* gw1    = (const float*)d_in[7];
    const float* gb1    = (const float*)d_in[8];
    const float* gw2    = (const float*)d_in[9];
    const float* gb2    = (const float*)d_in[10];
    const float* pln1_w = (const float*)d_in[11];
    const float* pln1_b = (const float*)d_in[12];
    const float* pw1    = (const float*)d_in[13];
    const float* pb1    = (const float*)d_in[14];
    const float* pw2    = (const float*)d_in[15];
    const float* pb2    = (const float*)d_in[16];
    const float* pln2_w = (const float*)d_in[17];
    const float* pln2_b = (const float*)d_in[18];
    const float* in_w   = (const float*)d_in[19];
    const float* xw     = (const float*)d_in[20];
    const float* dtw    = (const float*)d_in[21];
    const float* dtb    = (const float*)d_in[22];
    const float* Alog   = (const float*)d_in[23];
    const float* Dp     = (const float*)d_in[24];
    const float* outw   = (const float*)d_in[25];
    const float* sln_w  = (const float*)d_in[26];
    const float* sln_b  = (const float*)d_in[27];

    char* ws = (char*)d_ws;
    const size_t NREG = (size_t)BB * RR;               // 1024 regions
    float*    ws_region = (float*)(ws + 0);            // 1024*512 f32 (2 MB)
    float*    ws_u      = (float*)(ws + 2097152);      // 2 MB
    float*    ws_z      = (float*)(ws + 4194304);      // 2 MB
    float*    ws_delta  = (float*)(ws + 6291456);      // 2 MB
    float*    ws_ypart  = (float*)(ws + 8388608);      // NG * 2 MB = 8 MB
    float*    ws_B      = (float*)(ws + 16777216);     // 1024*48 f32
    float*    ws_C      = (float*)(ws + 16973824);
    _Float16* w1t       = (_Float16*)(ws + 17170432);  // 512*384 f16
    _Float16* rt16      = (_Float16*)(ws + 17563648);  // 1024*512 f16
    float*    partials  = (float*)(ws + 18612224);     // 1024*2

    float* out = (float*)d_out;

    prep_w1t_kernel<<<(DP * DR + 255) / 256, 256, 0, stream>>>(gw1, w1t);

    region_kernel<<<NREG, 256, 0, stream>>>(
        pt, pmask, lp, gln_w, gln_b, w1t, gb1, gw2, gb2,
        pln1_w, pln1_b, pw1, pb1, pw2, pb2, pln2_w, pln2_b,
        in_w, xw, dtw, dtb,
        ws_region, ws_u, ws_z, ws_delta, ws_B, ws_C);

    scan_kernel<<<(BB * DR * NG) / 256, 256, 0, stream>>>(
        ws_delta, ws_u, ws_B, ws_C, Alog, ws_ypart);

    out_kernel<<<NREG, 256, 0, stream>>>(
        ws_ypart, ws_u, ws_z, ws_region, Dp, outw, sln_w, sln_b, out, rt16);

    sim_kernel<<<BB * 16 * 16, 32, 0, stream>>>(rt16, nearm, farm, partials);

    loss_kernel<<<1, 256, 0, stream>>>(partials, nearm, farm, out + (size_t)NREG * DR);
}
